// StatefulBilinearAttention_39324720562583
// MI455X (gfx1250) — matrix operand order repaired and checked by reading, not compiled, not run on hardware
//
#include <hip/hip_runtime.h>
#include <hip/hip_bf16.h>

// ---------------------------------------------------------------------------
// Problem constants (from reference): B=8, Q=128, K=1024, QS=SS=KS=VS=512, H=256
// Outputs (concat): h [B,Q,512] | w_qsk [B,Q,1024] | norm_attn [B,Q,1024]
// ---------------------------------------------------------------------------
#define BB   8
#define QQ   128
#define KK   1024
#define FS   512     // QS == SS == KS == VS
#define HH   256
#define QT   4       // queries per scores-block

typedef __attribute__((ext_vector_type(16))) __bf16 v16bf;
typedef __attribute__((ext_vector_type(8)))  float  v8f;

static __device__ __forceinline__ float fast_tanh(float x) {
  // tanh(x) = 1 - 2/(exp(2x)+1); exp2-based, saturates cleanly at +-inf
  float e = __builtin_amdgcn_exp2f(x * 2.8853900817779268f); // 2*log2(e)
  return 1.0f - 2.0f * __builtin_amdgcn_rcpf(e + 1.0f);
}
static __device__ __forceinline__ float fast_exp(float x) {
  return __builtin_amdgcn_exp2f(x * 1.4426950408889634f);
}

// ---------------------------------------------------------------------------
// Pre-swizzle an fp32 [Kdim x N] row-major matrix into bf16 WMMA B-panels.
// Panel tile (kt,nt) covers K=[kt*32,kt*32+32), N=[nt*16,nt*16+16).
// B-layout used here (16-bit, 32x16, same k order as A): lane L holds column N0+(L&15); half-wave
// selects a K-block of 16; lane's 16 bf16 are K = K0+kblk+0..15, stored
// contiguously -> one 32B vector load per lane per WMMA in the GEMM.
// ---------------------------------------------------------------------------
__global__ __launch_bounds__(256) void panelize_bf16_kernel(
    const float* __restrict__ W, __bf16* __restrict__ panel,
    int Kdim, int N, long long srcStride, long long dstStride)
{
  const int b  = blockIdx.y;
  const float* Wb = W + (size_t)b * srcStride;
  __bf16*      Pb = panel + (size_t)b * dstStride;

  const int KT = Kdim >> 5, NT = N >> 4;
  const int tile = blockIdx.x * (blockDim.x >> 5) + (threadIdx.x >> 5);
  if (tile >= KT * NT) return;
  const int lane = threadIdx.x & 31;
  const int nt = tile / KT, kt = tile % KT;
  const int kblk = (lane >> 4) << 4;           // 0 or 16
  const int n    = (nt << 4) + (lane & 15);
  const int K0   = kt << 5;

  v16bf t;
#pragma unroll
  for (int i = 0; i < 16; ++i)
    t[i] = (__bf16)Wb[(size_t)(K0 + kblk + i) * N + n];

  *(v16bf*)(Pb + (((size_t)(nt * KT + kt)) * 32 + lane) * 16) = t;
}

// ---------------------------------------------------------------------------
// WMMA bf16 GEMM, N-blocked: each wave computes a 16x64 C strip (4 N-tiles),
// so each A fragment (loaded fp32, cvt to bf16) feeds 4 v_wmma ops.
// C = A0*P0 (+ A1*P1) (+ bias); fp32 accumulate. Batched via blockIdx.y.
// ---------------------------------------------------------------------------
__global__ __launch_bounds__(256) void wmma_gemm_kernel(
    const float* __restrict__ A0, const float* __restrict__ A1,
    const __bf16* __restrict__ P0, const __bf16* __restrict__ P1,
    const float* __restrict__ bias, float* __restrict__ C,
    int M, int Kdim, int N, long long sA, long long sP, long long sC)
{
  const int b   = blockIdx.y;
  const int KT  = Kdim >> 5, NT = N >> 4, MT = M >> 4;
  const int NT4 = NT >> 2;
  const int wv  = blockIdx.x * (blockDim.x >> 5) + (threadIdx.x >> 5);
  if (wv >= MT * NT4) return;
  const int lane = threadIdx.x & 31;
  const int mt  = wv / NT4;
  const int ntb = (wv % NT4) << 2;             // first of 4 adjacent N-tiles
  const int row = (mt << 4) + (lane & 15);     // A-matrix row for this lane
  const int kb  = (lane >> 4) << 4;            // A K-subblock base: 0 or 16

  v8f c[4];
#pragma unroll
  for (int t = 0; t < 4; ++t) c[t] = v8f{0.f,0.f,0.f,0.f,0.f,0.f,0.f,0.f};

  const float*  Alist[2] = {A0, A1};
  const __bf16* Plist[2] = {P0, P1};

  for (int ph = 0; ph < 2; ++ph) {
    const float* A = Alist[ph];
    if (!A) break;
    const float*  arow = A + (size_t)b * sA + (size_t)row * Kdim;
    const __bf16* P    = Plist[ph] + (size_t)b * sP;
    for (int kt = 0; kt < KT; ++kt) {
      // A fragment: K = K0+kb+{0..15} (one contiguous run)
      const float* ap = arow + (kt << 5) + kb;
      float4 f0 = *(const float4*)(ap);
      float4 f1 = *(const float4*)(ap + 4);
      float4 f2 = *(const float4*)(ap + 8);
      float4 f3 = *(const float4*)(ap + 12);
      v16bf a;
      a[0]  = (__bf16)f0.x; a[1]  = (__bf16)f0.y; a[2]  = (__bf16)f0.z; a[3]  = (__bf16)f0.w;
      a[4]  = (__bf16)f1.x; a[5]  = (__bf16)f1.y; a[6]  = (__bf16)f1.z; a[7]  = (__bf16)f1.w;
      a[8]  = (__bf16)f2.x; a[9]  = (__bf16)f2.y; a[10] = (__bf16)f2.z; a[11] = (__bf16)f2.w;
      a[12] = (__bf16)f3.x; a[13] = (__bf16)f3.y; a[14] = (__bf16)f3.z; a[15] = (__bf16)f3.w;

#pragma unroll
      for (int t = 0; t < 4; ++t) {
        v16bf bm = *(const v16bf*)(P + (((size_t)((ntb + t) * KT + kt)) * 32 + lane) * 16);
        c[t] = __builtin_amdgcn_wmma_f32_16x16x32_bf16(
                   /*neg_a=*/false, a, /*neg_b=*/false, bm,
                   /*c_mod=*/(short)0, c[t], /*reuse_a=*/false, /*reuse_b=*/false);
      }
    }
  }

  // C layout: lane holds column n; VGPR j -> row m0+j (m0 = +8 for upper half-wave)
  float* Cb = C + (size_t)b * sC;
  const int m0 = (mt << 4) + ((lane >> 4) << 3);
#pragma unroll
  for (int t = 0; t < 4; ++t) {
    const int n = ((ntb + t) << 4) + (lane & 15);
    const float bv = bias ? bias[n] : 0.0f;
#pragma unroll
    for (int j = 0; j < 8; ++j)
      Cb[(size_t)(m0 + j) * N + n] = c[t][j] + bv;
  }
}

// ---------------------------------------------------------------------------
// Scores + softmax, QT=4 queries per block: each wk row chunk loaded once is
// reused for 4 queries (4x less L2 traffic; tanh count is irreducible).
// One block per (b, q-group): 8 waves; wave owns k = wave, wave+8, ...
// Lane handles an 8-wide H chunk -> 8 tanh+FMA per query, butterfly reduce.
// Mask + w_qsk stores done in a coalesced LDS pass, then row softmax per q.
// ---------------------------------------------------------------------------
__global__ __launch_bounds__(256) void scores_softmax_kernel(
    const float* __restrict__ aq,   // [B*Q, 256]
    const float* __restrict__ wk,   // [B*K, 256]
    const float* __restrict__ vvec, // [256]
    const unsigned char* __restrict__ mask, // [B*Q, 1024] bool
    float* __restrict__ w_out,      // [B*Q, 1024]
    float* __restrict__ p_out)      // [B*Q, 1024]
{
  const int bq0  = blockIdx.x * QT;   // first (b,q) row of this block
  const int b    = bq0 >> 7;          // Q = 128 (QT divides Q, so same b)
  const int tid  = threadIdx.x;
  const int wave = tid >> 5;
  const int lane = tid & 31;
  const float NEG_INF = -__builtin_inff();

  __shared__ float aq_s[QT][HH];
  __shared__ float v_s[HH];
  __shared__ float sc[QT][KK];
  __shared__ float red[16];

  for (int i = tid; i < QT * HH; i += 256)
    ((float*)aq_s)[i] = aq[(size_t)bq0 * HH + i];
  v_s[tid] = vvec[tid];
  __syncthreads();

  // cache this lane's H-chunk for all QT queries in registers
  const int h0 = lane * 8;
  float aqr[QT][8], vr[8];
#pragma unroll
  for (int j = 0; j < 8; ++j) vr[j] = v_s[h0 + j];
#pragma unroll
  for (int q = 0; q < QT; ++q)
#pragma unroll
    for (int j = 0; j < 8; ++j) aqr[q][j] = aq_s[q][h0 + j];

  const float* wkb = wk + (size_t)b * KK * HH;

  for (int k = wave; k < KK; k += 8) {
    const float* wrow = wkb + (size_t)k * HH + h0;
    float4 g0 = *(const float4*)(wrow);
    float4 g1 = *(const float4*)(wrow + 4);
    float g[8] = {g0.x, g0.y, g0.z, g0.w, g1.x, g1.y, g1.z, g1.w};
#pragma unroll
    for (int q = 0; q < QT; ++q) {
      float s = 0.0f;
#pragma unroll
      for (int j = 0; j < 8; ++j)
        s += fast_tanh(aqr[q][j] + g[j]) * vr[j];
#pragma unroll
      for (int off = 16; off; off >>= 1) s += __shfl_xor(s, off, 32);
      if (lane == 0) sc[q][k] = s;
    }
  }
  __syncthreads();

  // coalesced mask apply + w_qsk store
  for (int q = 0; q < QT; ++q) {
    const size_t rowoff = (size_t)(bq0 + q) * KK;
    for (int k = tid; k < KK; k += 256) {
      float val = mask[rowoff + k] ? sc[q][k] : NEG_INF;
      sc[q][k] = val;
      w_out[rowoff + k] = val;
    }
  }
  __syncthreads();

  // per-query row softmax over K=1024
  for (int q = 0; q < QT; ++q) {
    float lmax = NEG_INF;
    for (int k = tid; k < KK; k += 256) lmax = fmaxf(lmax, sc[q][k]);
#pragma unroll
    for (int off = 16; off; off >>= 1) lmax = fmaxf(lmax, __shfl_xor(lmax, off, 32));
    if (lane == 0) red[wave] = lmax;
    __syncthreads();
    if (tid == 0) {
      float m = red[0];
      for (int i = 1; i < 8; ++i) m = fmaxf(m, red[i]);
      red[0] = m;
    }
    __syncthreads();
    const float mx = red[0];

    float lsum = 0.0f;
    for (int k = tid; k < KK; k += 256) {
      float e = fast_exp(sc[q][k] - mx);   // exp(-inf) -> 0 for masked entries
      sc[q][k] = e;
      lsum += e;
    }
#pragma unroll
    for (int off = 16; off; off >>= 1) lsum += __shfl_xor(lsum, off, 32);
    if (lane == 0) red[8 + wave] = lsum;
    __syncthreads();
    if (tid == 0) {
      float t = 0.0f;
      for (int i = 0; i < 8; ++i) t += red[8 + i];
      red[8] = t;
    }
    __syncthreads();
    const float inv = 1.0f / red[8];

    const size_t rowoff = (size_t)(bq0 + q) * KK;
    for (int k = tid; k < KK; k += 256)
      p_out[rowoff + k] = sc[q][k] * inv;   // masked -> 0 automatically
    __syncthreads();
  }
}

// ---------------------------------------------------------------------------
// Host-side launcher
// ---------------------------------------------------------------------------
extern "C" void kernel_launch(void* const* d_in, const int* in_sizes, int n_in,
                              void* d_out, int out_size, void* d_ws, size_t ws_size,
                              hipStream_t stream) {
  (void)in_sizes; (void)n_in; (void)out_size; (void)ws_size;

  const float* query = (const float*)d_in[0];   // [B,Q,512]
  const float* state = (const float*)d_in[1];   // [B,Q,512]
  const float* keyi  = (const float*)d_in[2];   // [B,K,512]
  const float* value = (const float*)d_in[3];   // [B,K,512]
  const unsigned char* mask = (const unsigned char*)d_in[4]; // [B,Q,K] bool
  const float* Wq    = (const float*)d_in[5];   // [512,256]
  const float* bq    = (const float*)d_in[6];   // [256]
  const float* Ws    = (const float*)d_in[7];   // [512,256]
  const float* Wk    = (const float*)d_in[8];   // [512,256]
  const float* vvec  = (const float*)d_in[9];   // [256]

  float* out   = (float*)d_out;
  float* h_out = out;                                   // [B,Q,512]
  float* w_out = out + (size_t)BB * QQ * FS;            // [B,Q,1024]
  float* p_out = w_out + (size_t)BB * QQ * KK;          // [B,Q,1024]

  // workspace layout (bytes)
  char* wsb = (char*)d_ws;
  float*  ws_aq = (float*)(wsb + 0);                       // [1024,256] f32, 1 MB
  float*  ws_wk = (float*)(wsb + (1u << 20));              // [8192,256] f32, 8 MB
  __bf16* pWq   = (__bf16*)(wsb + 9437184);                // 256 KB
  __bf16* pWs   = (__bf16*)(wsb + 9699328);                // 256 KB
  __bf16* pWk   = (__bf16*)(wsb + 9961472);                // 256 KB
  __bf16* pV    = (__bf16*)(wsb + 10223616);               // [B][1024,512] bf16, 8 MB

  // 1) panelize weights: [512x256] -> 256 tiles -> 32 blocks of 8 waves
  panelize_bf16_kernel<<<dim3(32, 1), 256, 0, stream>>>(Wq, pWq, FS, HH, 0, 0);
  panelize_bf16_kernel<<<dim3(32, 1), 256, 0, stream>>>(Ws, pWs, FS, HH, 0, 0);
  panelize_bf16_kernel<<<dim3(32, 1), 256, 0, stream>>>(Wk, pWk, FS, HH, 0, 0);
  // panelize value per batch: [1024x512] -> 1024 tiles -> 128 blocks, y = batch
  panelize_bf16_kernel<<<dim3(128, BB), 256, 0, stream>>>(
      value, pV, KK, FS, (long long)KK * FS, (long long)KK * FS);

  // 2) AQ = query@Wq + state@Ws + bq : M=1024, K=512, N=256 -> 64*4 strips
  wmma_gemm_kernel<<<dim3(32, 1), 256, 0, stream>>>(
      query, state, pWq, pWs, bq, ws_aq, BB * QQ, FS, HH, 0, 0, 0);

  // 3) WK = key@Wk : M=8192, K=512, N=256 -> 512*4 strips
  wmma_gemm_kernel<<<dim3(256, 1), 256, 0, stream>>>(
      keyi, nullptr, pWk, nullptr, nullptr, ws_wk, BB * KK, FS, HH, 0, 0, 0);

  // 4) scores + softmax: one block per (b, 4 queries)
  scores_softmax_kernel<<<dim3(BB * QQ / QT, 1), 256, 0, stream>>>(
      ws_aq, ws_wk, vvec, mask, w_out, p_out);

  // 5) h = norm_attn @ value : batched, M=128, K=1024, N=512 -> 8*8 strips
  wmma_gemm_kernel<<<dim3(8, BB), 256, 0, stream>>>(
      p_out, nullptr, pV, nullptr, nullptr, h_out, QQ, KK, FS,
      (long long)QQ * KK, (long long)KK * FS, (long long)QQ * FS);
}
